// LinearAndSoftmax_38319698215661
// MI455X (gfx1250) — compile-verified, matrix operand order checked by reading, not run on hardware
//
#include <hip/hip_runtime.h>

#define D_MODEL 768
#define VOCAB   30523
#define VPAD    30528      // pad vocab to multiple of 32
#define NCHUNK  (VPAD / 32)
#define NTOK    8192       // 16*512 tokens
#define MTILE   64         // tokens per workgroup
#define CHUNK   32         // vocab columns per chunk
#define KTILE   128        // K staged into LDS per step
#define NKO     (D_MODEL / KTILE)   // 6
#define XS_STRIDE 136      // 128 + 8 pad (ushort units) -> conflict-free ds_load_b128
#define WS_STRIDE 136

typedef __attribute__((ext_vector_type(16))) __bf16 v16bf;
typedef __attribute__((ext_vector_type(8)))  float  v8f;
typedef __attribute__((ext_vector_type(4)))  unsigned int v4u;
typedef __attribute__((ext_vector_type(4)))  int    v4i;

union FragBF { v4u u[2]; v16bf v; };

// AS-qualified pointer types for the async-DMA builtin
// (clang prints addrspace(1) as `__device__`, addrspace(3) as `__shared__`)
typedef __attribute__((address_space(1))) v4i* gv4i_p;   // global
typedef __attribute__((address_space(3))) v4i* lv4i_p;   // LDS

// async DMA: global -> LDS, 16 bytes per lane, tracked by ASYNCcnt
__device__ __forceinline__ void async_ld_b128(const unsigned short* g, unsigned short* l) {
  __builtin_amdgcn_global_load_async_to_lds_b128((gv4i_p)(const void*)g, (lv4i_p)(void*)l, 0, 0);
}

__device__ __forceinline__ unsigned short f2bf(float f) {
  unsigned int u = __builtin_bit_cast(unsigned int, f);
  u += 0x7FFFu + ((u >> 16) & 1u);          // round-to-nearest-even
  return (unsigned short)(u >> 16);
}

// ---- prologue: fp32 -> bf16 conversion (zero-fills the vocab pad rows) ----
__global__ void cvt_bf16_kernel(const float* __restrict__ src,
                                unsigned short* __restrict__ dst,
                                long n_src, long n_total) {
  long i = (long)blockIdx.x * blockDim.x + threadIdx.x;
  long stride = (long)gridDim.x * blockDim.x;
  for (; i < n_total; i += stride)
    dst[i] = (i < n_src) ? f2bf(src[i]) : (unsigned short)0;
}

// ---- fused GEMM + online log-softmax + NLL gather ----
__global__ __launch_bounds__(256)
void fused_ce_kernel(const unsigned short* __restrict__ xb,   // [NTOK][768] bf16
                     const unsigned short* __restrict__ Wb,   // [VPAD][768] bf16
                     const float* __restrict__ bias,          // [VOCAB]
                     const int*   __restrict__ label,         // [NTOK]
                     float*       __restrict__ nll_out) {     // [NTOK]
  __shared__ unsigned short xsb[2][MTILE * XS_STRIDE];   // 2 x 17408 B
  __shared__ unsigned short wsb[2][CHUNK * WS_STRIDE];   // 2 x  8704 B
  __shared__ float redM[2][MTILE], redS[2][MTILE], redL[2][MTILE];

  const int tid  = threadIdx.x;
  const int lane = tid & 31;
  const int wid  = tid >> 5;
  const int mrow = wid & 3;    // 4 wave-rows over tokens
  const int vcol = wid >> 2;   // 2 wave-cols over vocab chunk
  const int lr   = lane & 15;
  const int lh   = lane >> 4;

  const int tokBase = blockIdx.x * MTILE;

  // issue one K-tile's async loads (x: 4 b128/thread, W: 2 b128/thread)
  auto issue_tile = [&](int ko, int chunkIdx, int buf) {
#pragma unroll
    for (int j = 0; j < 4; ++j) {
      int i = tid + j * 256, row = i >> 4, c = i & 15;
      async_ld_b128(xb + (size_t)(tokBase + row) * D_MODEL + ko * KTILE + c * 8,
                    &xsb[buf][row * XS_STRIDE + c * 8]);
    }
#pragma unroll
    for (int j = 0; j < 2; ++j) {
      int i = tid + j * 256, row = i >> 4, c = i & 15;
      async_ld_b128(Wb + (size_t)(chunkIdx * CHUNK + row) * D_MODEL + ko * KTILE + c * 8,
                    &wsb[buf][row * WS_STRIDE + c * 8]);
    }
  };

  // labels for this lane's 8 token rows (VGPR r -> row mrow*16 + lh*8 + r)
  int mylab[8];
#pragma unroll
  for (int r = 0; r < 8; ++r)
    mylab[r] = label[tokBase + mrow * 16 + lh * 8 + r];

  float m[8], s[8], lab[8];
#pragma unroll
  for (int r = 0; r < 8; ++r) { m[r] = -__builtin_inff(); s[r] = 0.f; lab[r] = 0.f; }

  // pipeline prologue: tile (chunk 0, ko 0) into buffer 0
  int p = 0;
  issue_tile(0, 0, 0);
  __builtin_amdgcn_s_wait_asynccnt(0);
  __syncthreads();

  for (int chunk = 0; chunk < NCHUNK; ++chunk) {
    v8f acc = {0.f, 0.f, 0.f, 0.f, 0.f, 0.f, 0.f, 0.f};

    for (int ko = 0; ko < NKO; ++ko) {
      // prefetch next K-tile (clamped cross-chunk) into the other buffer;
      // safe: trailing barrier of previous iteration ordered all reads of it
      int nko = ko + 1, nchunk = chunk;
      if (nko == NKO) { nko = 0; nchunk = (chunk + 1 < NCHUNK) ? chunk + 1 : chunk; }
      issue_tile(nko, nchunk, p ^ 1);

      // only the freshly issued 6 loads may remain outstanding -> current tile done
      __builtin_amdgcn_s_wait_asynccnt(6);
      __syncthreads();

      // 4 x (16x16x32 bf16) WMMA against resident tile
#pragma unroll
      for (int kk = 0; kk < KTILE / 32; ++kk) {
        // A fragment: lanes 0-15 hold K 0-7 & 16-23, lanes 16-31 hold K 8-15 & 24-31
        const unsigned short* xrow = &xsb[p][(mrow * 16 + lr) * XS_STRIDE + kk * 32 + lh * 8];
        FragBF A; A.u[0] = *(const v4u*)(xrow); A.u[1] = *(const v4u*)(xrow + 16);
        // B fragment of W^T: identical pattern over W rows (col n of B == row n of W)
        const unsigned short* wrow = &wsb[p][(vcol * 16 + lr) * WS_STRIDE + kk * 32 + lh * 8];
        FragBF B; B.u[0] = *(const v4u*)(wrow); B.u[1] = *(const v4u*)(wrow + 16);
        acc = __builtin_amdgcn_wmma_f32_16x16x32_bf16(false, A.v, false, B.v,
                                                      (short)0, acc, false, false);
      }
      __syncthreads();   // all reads of buffer p done -> it may be refilled
      p ^= 1;
    }

    // online logsumexp update: per (r,lane) exactly one new logit this chunk
    const int  col   = chunk * CHUNK + vcol * 16 + lr;
    const bool valid = (col < VOCAB);
    const float bv   = valid ? bias[col] : 0.f;
#pragma unroll
    for (int r = 0; r < 8; ++r) {
      float v = acc[r] + bv;
      v = valid ? v : -__builtin_inff();
      float nm = fmaxf(m[r], v);
      s[r] = s[r] * __expf(m[r] - nm) + __expf(v - nm);
      m[r] = nm;
      if (col == mylab[r]) lab[r] = v;   // exactly one (lane,vcol) matches per row
    }
  }

  // butterfly merge across the 16 lanes of each half (masks stay within halves)
#pragma unroll
  for (int k = 1; k <= 8; k <<= 1) {
#pragma unroll
    for (int r = 0; r < 8; ++r) {
      float om = __shfl_xor(m[r], k, 32);
      float os = __shfl_xor(s[r], k, 32);
      float ol = __shfl_xor(lab[r], k, 32);
      float nm = fmaxf(m[r], om);
      s[r] = s[r] * __expf(m[r] - nm) + os * __expf(om - nm);
      m[r] = nm;
      lab[r] += ol;
    }
  }
  if (lr == 0) {
#pragma unroll
    for (int r = 0; r < 8; ++r) {
      int rowl = mrow * 16 + lh * 8 + r;
      redM[vcol][rowl] = m[r]; redS[vcol][rowl] = s[r]; redL[vcol][rowl] = lab[r];
    }
  }
  __syncthreads();

  // merge the two vocab wave-columns, emit per-token NLL
  if (tid < MTILE) {
    float m0 = redM[0][tid], m1 = redM[1][tid];
    float M  = fmaxf(m0, m1);
    float S  = redS[0][tid] * __expf(m0 - M) + redS[1][tid] * __expf(m1 - M);
    float L  = redL[0][tid] + redL[1][tid];
    nll_out[tokBase + tid] = (logf(S) + M) - L;   // logsumexp - logit[label]
  }
}

// ---- epilogue: mean over 8192 NLLs -> scalar ----
__global__ void reduce_mean_kernel(const float* __restrict__ nll,
                                   float* __restrict__ out) {
  __shared__ float sm[256];
  float acc = 0.f;
  for (int i = threadIdx.x; i < NTOK; i += 256) acc += nll[i];
  sm[threadIdx.x] = acc;
  __syncthreads();
  for (int off = 128; off > 0; off >>= 1) {
    if ((int)threadIdx.x < off) sm[threadIdx.x] += sm[threadIdx.x + off];
    __syncthreads();
  }
  if (threadIdx.x == 0) out[0] = sm[0] / (float)NTOK;
}

extern "C" void kernel_launch(void* const* d_in, const int* in_sizes, int n_in,
                              void* d_out, int out_size, void* d_ws, size_t ws_size,
                              hipStream_t stream) {
  const float* x     = (const float*)d_in[0];   // [16,512,768]
  const float* W     = (const float*)d_in[1];   // [30523,768]
  const float* b     = (const float*)d_in[2];   // [30523]
  const int*   label = (const int*)d_in[3];     // [16,512]

  // workspace layout: W_bf16 [VPAD*768] | x_bf16 [NTOK*768] | nll [NTOK] floats
  unsigned short* Wb  = (unsigned short*)d_ws;
  unsigned short* xbw = Wb + (size_t)VPAD * D_MODEL;
  float*          nll = (float*)(xbw + (size_t)NTOK * D_MODEL);

  {
    long n_src = (long)VOCAB * D_MODEL, n_tot = (long)VPAD * D_MODEL;
    int blocks = (int)((n_tot + 255) / 256);
    cvt_bf16_kernel<<<blocks, 256, 0, stream>>>(W, Wb, n_src, n_tot);
  }
  {
    long n = (long)NTOK * D_MODEL;
    cvt_bf16_kernel<<<(int)((n + 255) / 256), 256, 0, stream>>>(x, xbw, n, n);
  }
  fused_ce_kernel<<<NTOK / MTILE, 256, 0, stream>>>(xbw, Wb, b, label, nll);
  reduce_mean_kernel<<<1, 256, 0, stream>>>(nll, (float*)d_out);
}
